// MSA_10969346474559
// MI455X (gfx1250) — matrix operand order repaired and checked by reading, not compiled, not run on hardware
//
#include <hip/hip_runtime.h>
#include <hip/hip_bf16.h>

// MI455X flash-attention: bf16 WMMA (v_wmma_f32_16x16x32_bf16), f32 accumulate,
// async global->LDS double-buffered K/V staging.
// B=4, S=2048, D=1024, H=16, DH=64.

#define Bc 4
#define Sc 2048
#define Dc 1024
#define Hc 16
#define DHc 64

typedef __bf16 bf16_t;
typedef __attribute__((ext_vector_type(8)))  __bf16 v8bf;
typedef __attribute__((ext_vector_type(16))) __bf16 v16bf;
typedef __attribute__((ext_vector_type(8)))  float  v8f;

#define WMMA_BF16(A, B, C) \
  __builtin_amdgcn_wmma_f32_16x16x32_bf16(false, (A), false, (B), (short)0, (C), false, false)

// ---------- async global->LDS copy (gfx1250), with sync fallback -----------
#if defined(__has_builtin)
#if __has_builtin(__builtin_amdgcn_global_load_async_to_lds_b128)
#define HAVE_ASYNC_LDS 1
#endif
#endif

typedef int v4i_gcc __attribute__((vector_size(16)));
typedef __attribute__((address_space(1))) v4i_gcc as1_v4i;
typedef __attribute__((address_space(3))) v4i_gcc as3_v4i;

static __device__ inline void cp16_g2lds(bf16_t* lds, const bf16_t* g) {
#ifdef HAVE_ASYNC_LDS
  // AS1 pointers share the flat representation; AS3 pointer = low 32 bits of
  // the flat LDS address (ISA 10.2 aperture rule).
  __builtin_amdgcn_global_load_async_to_lds_b128(
      (as1_v4i*)(unsigned long long)g,
      (as3_v4i*)(unsigned int)(unsigned long long)lds, 0, 0);
#else
  *(v8bf*)lds = *(const v8bf*)g;
#endif
}

#ifdef HAVE_ASYNC_LDS
#if __has_builtin(__builtin_amdgcn_s_wait_asynccnt)
#define WAIT_ASYNC(n) __builtin_amdgcn_s_wait_asynccnt(n)
#else
#define WAIT_ASYNC(n) asm volatile("s_wait_asynccnt %0" ::"i"(n) : "memory")
#endif
#else
#define WAIT_ASYNC(n) ((void)0)
#endif

// ---- A-matrix (16x32 bf16) load from row-major LDS tile -------------------
// ISA: lane L holds row M=L%16; elems 0..7 = K base..base+7, elems 8..15 =
// K base+16..base+23, base = (L<16 ? 0 : 8).
static __device__ inline v16bf load_A16x32(const bf16_t* base, int strideElems, int lane) {
  const int M  = lane & 15;
  const int hi = (lane >> 4) & 1;
  const bf16_t* p = base + M * strideElems + hi * 8;
  v8bf lo = *(const v8bf*)(p);
  v8bf h2 = *(const v8bf*)(p + 16);
  return __builtin_shufflevector(lo, h2, 0,1,2,3,4,5,6,7,8,9,10,11,12,13,14,15);
}

// ---- B-matrix (32x16 bf16) load: stored row-major [N][K] ------------------
// ISA: lane L holds column N=L%16; elems 0..7 = K base..base+7, elems 8..15 =
// K base+16..base+23, base = (L<16 ? 0 : 8).
static __device__ inline v16bf load_B32x16(const bf16_t* base, int strideElems, int lane) {
  const int N  = lane & 15;
  const int kh = (lane >> 4) & 1;
  const bf16_t* p = base + N * strideElems + kh * 8;
  v8bf lo = *(const v8bf*)(p);
  v8bf hi = *(const v8bf*)(p + 16);
  return __builtin_shufflevector(lo, hi, 0,1,2,3,4,5,6,7,8,9,10,11,12,13,14,15);
}

// ===========================================================================
// Kernel 1: per-head QKV projection, f32 in -> bf16 out.
// q (pre-scaled by (1/sqrt(DH))*log2(e)), k: [B,H,S,DH]; v^T: [B,H,DH,S]
// ===========================================================================
__global__ __launch_bounds__(256) void msa_qkv_kernel(
    const float* __restrict__ x,
    const float* __restrict__ Wq, const float* __restrict__ bq,
    const float* __restrict__ Wk, const float* __restrict__ bk,
    const float* __restrict__ Wv, const float* __restrict__ bv,
    bf16_t* __restrict__ qo, bf16_t* __restrict__ ko, bf16_t* __restrict__ vto) {
  const int bh = blockIdx.y;
  const int b  = bh >> 4, h = bh & 15;
  const int sb = blockIdx.x * 32;
  const int tid = threadIdx.x;

  __shared__ float Wl[3 * 64 * 65];   // +1 pad: conflict-free column reads
  __shared__ float xs[32 * 64];

  for (int m = 0; m < 3; ++m) {
    const float* Wg = (m == 0 ? Wq : (m == 1 ? Wk : Wv)) + h * 4096;
    float* Wd = Wl + m * 64 * 65;
#pragma unroll
    for (int i = 0; i < 16; ++i) {
      int idx = i * 256 + tid;
      int e = idx >> 6, d = idx & 63;
      Wd[e * 65 + d] = Wg[idx];
    }
  }
#pragma unroll
  for (int i = 0; i < 8; ++i) {
    int idx = i * 256 + tid;
    int s = idx >> 6, d = idx & 63;
    xs[idx] = x[(long long)(b * Sc + sb + s) * Dc + h * DHc + d];
  }
  __syncthreads();

  const int e  = tid & 63;
  const int sg = tid >> 6;
  const float qscale = 0.18033688011112042f;  // (1/sqrt(64)) * log2(e)
  const float biasq = bq[h * 64 + e];
  const float biask = bk[h * 64 + e];
  const float biasv = bv[h * 64 + e];
  const float* wq = Wl + e * 65;
  const float* wk = Wl + 64 * 65 + e * 65;
  const float* wv = Wl + 2 * 64 * 65 + e * 65;

  for (int j = 0; j < 8; ++j) {
    const int s = sg * 8 + j;
    const float* xr = xs + s * 64;
    float aq = biasq, ak = biask, av = biasv;
#pragma unroll
    for (int d = 0; d < 64; ++d) {
      float xv = xr[d];
      aq = fmaf(wq[d], xv, aq);
      ak = fmaf(wk[d], xv, ak);
      av = fmaf(wv[d], xv, av);
    }
    const long long o = (long long)(bh * Sc + sb + s) * DHc + e;
    qo[o] = (bf16_t)(aq * qscale);
    ko[o] = (bf16_t)ak;
    vto[(long long)(bh * DHc + e) * Sc + sb + s] = (bf16_t)av;
  }
}

// ===========================================================================
// Kernel 2: flash attention. 8 waves/block; wave owns 16 query rows;
// 64-key tiles, double-buffered async K/V^T staging.
// ===========================================================================
#define TSTR 72                     // padded LDS row stride (bf16 elems)
#define NKB  (Sc / 64)              // 32 key blocks

__global__ __launch_bounds__(256) void msa_flash_kernel(
    const bf16_t* __restrict__ qg, const bf16_t* __restrict__ kg,
    const bf16_t* __restrict__ vtg, float* __restrict__ out) {
  const int bh   = blockIdx.y;
  const int b    = bh >> 4, h = bh & 15;
  const int tid  = threadIdx.x;
  const int wave = tid >> 5, lane = tid & 31;

  __shared__ bf16_t QPs[8 * 16 * TSTR];      // Q staging, then P scratch
  __shared__ bf16_t Ks[2][64 * TSTR];        // double-buffered K tiles
  __shared__ bf16_t Vs[2][64 * TSTR];        // double-buffered V^T tiles

  const long long baseQK = (long long)bh * Sc * DHc;
  const long long baseVT = (long long)bh * DHc * Sc;
  const int qbase = blockIdx.x * 128 + wave * 16;

  // ---- stage this wave's 16x64 Q tile, build A operands, region -> P ------
  bf16_t* qls = QPs + wave * 16 * TSTR;
#pragma unroll
  for (int i = 0; i < 4; ++i) {
    int chunk = i * 32 + lane;
    int r = chunk >> 3, c = chunk & 7;
    v8bf v = *(const v8bf*)(qg + baseQK + (long long)(qbase + r) * DHc + c * 8);
    *(v8bf*)(qls + r * TSTR + c * 8) = v;
  }
  const v16bf aq0 = load_A16x32(qls + 0,  TSTR, lane);   // dh 0..31
  const v16bf aq1 = load_A16x32(qls + 32, TSTR, lane);   // dh 32..63

  // ---- tile staging: 64 rows x 64 cols bf16, 2 x b128 per thread each -----
  const int srow = tid >> 3, scol = (tid & 7) * 8;   // chunks 0..255
#define STAGE(kb, buf)                                                        \
  do {                                                                        \
    _Pragma("unroll")                                                         \
    for (int i = 0; i < 2; ++i) {                                             \
      int rr = srow + i * 32;                                                 \
      cp16_g2lds(&Ks[buf][rr * TSTR + scol],                                  \
                 kg + baseQK + (long long)((kb) * 64 + rr) * DHc + scol);     \
      cp16_g2lds(&Vs[buf][rr * TSTR + scol],                                  \
                 vtg + baseVT + (long long)rr * Sc + (kb) * 64 + scol);       \
    }                                                                         \
  } while (0)

  STAGE(0, 0);

  v8f O0 = {}, O1 = {}, O2 = {}, O3 = {};
  float m[8], l[8];
#pragma unroll
  for (int r = 0; r < 8; ++r) { m[r] = -1e30f; l[r] = 0.0f; }

  bf16_t* pls = qls;                          // reuse Q region as P scratch
  const int rowoff = (lane >> 4) * 8;
  const int ncol   = lane & 15;

  for (int kb = 0; kb < NKB; ++kb) {
    const int cur = kb & 1;
    if (kb + 1 < NKB) {
      STAGE(kb + 1, cur ^ 1);
      WAIT_ASYNC(4);                          // current tile's 4 ops done
    } else {
      WAIT_ASYNC(0);
    }
    __syncthreads();
    const bf16_t* Kc = Ks[cur];
    const bf16_t* Vc = Vs[cur];

    // ---- scores: 4 key sub-tiles x 2 chained dh chunks = 8 WMMAs ----------
    v8f s0 = {}, s1 = {}, s2 = {}, s3 = {};
    s0 = WMMA_BF16(aq0, load_B32x16(Kc + 0  * TSTR + 0, TSTR, lane), s0);
    s1 = WMMA_BF16(aq0, load_B32x16(Kc + 16 * TSTR + 0, TSTR, lane), s1);
    s2 = WMMA_BF16(aq0, load_B32x16(Kc + 32 * TSTR + 0, TSTR, lane), s2);
    s3 = WMMA_BF16(aq0, load_B32x16(Kc + 48 * TSTR + 0, TSTR, lane), s3);
    s0 = WMMA_BF16(aq1, load_B32x16(Kc + 0  * TSTR + 32, TSTR, lane), s0);
    s1 = WMMA_BF16(aq1, load_B32x16(Kc + 16 * TSTR + 32, TSTR, lane), s1);
    s2 = WMMA_BF16(aq1, load_B32x16(Kc + 32 * TSTR + 32, TSTR, lane), s2);
    s3 = WMMA_BF16(aq1, load_B32x16(Kc + 48 * TSTR + 32, TSTR, lane), s3);

    // ---- online softmax (scores already in base-2 domain) -----------------
    float t[8];
#pragma unroll
    for (int r = 0; r < 8; ++r)
      t[r] = fmaxf(fmaxf(s0[r], s1[r]), fmaxf(s2[r], s3[r]));
#pragma unroll
    for (int r = 0; r < 8; ++r) {             // row max across 16-lane half
      t[r] = fmaxf(t[r], __shfl_xor(t[r], 1, 32));
      t[r] = fmaxf(t[r], __shfl_xor(t[r], 2, 32));
      t[r] = fmaxf(t[r], __shfl_xor(t[r], 4, 32));
      t[r] = fmaxf(t[r], __shfl_xor(t[r], 8, 32));
    }
    float al[8], p0[8], p1[8], p2[8], p3[8];
#pragma unroll
    for (int r = 0; r < 8; ++r) {
      float mn = fmaxf(m[r], t[r]);
      al[r] = exp2f(m[r] - mn);
      m[r]  = mn;
      p0[r] = exp2f(s0[r] - mn);
      p1[r] = exp2f(s1[r] - mn);
      p2[r] = exp2f(s2[r] - mn);
      p3[r] = exp2f(s3[r] - mn);
    }
    float rs[8];
#pragma unroll
    for (int r = 0; r < 8; ++r) {
      rs[r] = (p0[r] + p1[r]) + (p2[r] + p3[r]);
      rs[r] += __shfl_xor(rs[r], 1, 32);
      rs[r] += __shfl_xor(rs[r], 2, 32);
      rs[r] += __shfl_xor(rs[r], 4, 32);
      rs[r] += __shfl_xor(rs[r], 8, 32);
      l[r] = l[r] * al[r] + rs[r];
    }
#pragma unroll
    for (int r = 0; r < 8; ++r) {
      O0[r] *= al[r]; O1[r] *= al[r]; O2[r] *= al[r]; O3[r] *= al[r];
    }

    // ---- P: C layout -> LDS -> A layout (16x64, two A tiles) --------------
#pragma unroll
    for (int r = 0; r < 8; ++r) {
      bf16_t* pr = pls + (r + rowoff) * TSTR + ncol;
      pr[0]  = (bf16_t)p0[r];
      pr[16] = (bf16_t)p1[r];
      pr[32] = (bf16_t)p2[r];
      pr[48] = (bf16_t)p3[r];
    }
    v16bf ap0 = load_A16x32(pls + 0,  TSTR, lane);   // keys 0..31
    v16bf ap1 = load_A16x32(pls + 32, TSTR, lane);   // keys 32..63

    // ---- O += P(16x64) . V(64x64): 8 WMMAs --------------------------------
    O0 = WMMA_BF16(ap0, load_B32x16(Vc + 0  * TSTR + 0, TSTR, lane), O0);
    O1 = WMMA_BF16(ap0, load_B32x16(Vc + 16 * TSTR + 0, TSTR, lane), O1);
    O2 = WMMA_BF16(ap0, load_B32x16(Vc + 32 * TSTR + 0, TSTR, lane), O2);
    O3 = WMMA_BF16(ap0, load_B32x16(Vc + 48 * TSTR + 0, TSTR, lane), O3);
    O0 = WMMA_BF16(ap1, load_B32x16(Vc + 0  * TSTR + 32, TSTR, lane), O0);
    O1 = WMMA_BF16(ap1, load_B32x16(Vc + 16 * TSTR + 32, TSTR, lane), O1);
    O2 = WMMA_BF16(ap1, load_B32x16(Vc + 32 * TSTR + 32, TSTR, lane), O2);
    O3 = WMMA_BF16(ap1, load_B32x16(Vc + 48 * TSTR + 32, TSTR, lane), O3);

    __syncthreads();                          // buffer reuse protection
  }

  // ---- epilogue: normalize and store [B,S,D] f32 --------------------------
#pragma unroll
  for (int r = 0; r < 8; ++r) {
    const float inv = 1.0f / l[r];
    const int q = qbase + r + rowoff;
    float* orow = out + (long long)(b * Sc + q) * Dc + h * DHc + ncol;
    orow[0]  = O0[r] * inv;
    orow[16] = O1[r] * inv;
    orow[32] = O2[r] * inv;
    orow[48] = O3[r] * inv;
  }
}

// ===========================================================================
extern "C" void kernel_launch(void* const* d_in, const int* in_sizes, int n_in,
                              void* d_out, int out_size, void* d_ws, size_t ws_size,
                              hipStream_t stream) {
  const float* x  = (const float*)d_in[0];
  const float* Wq = (const float*)d_in[1];
  const float* bq = (const float*)d_in[2];
  const float* Wk = (const float*)d_in[3];
  const float* bk = (const float*)d_in[4];
  const float* Wv = (const float*)d_in[5];
  const float* bv = (const float*)d_in[6];
  float* out = (float*)d_out;

  const size_t N = (size_t)Bc * Hc * Sc * DHc;
  bf16_t* qw  = (bf16_t*)d_ws;
  bf16_t* kw  = qw + N;
  bf16_t* vtw = kw + N;

  msa_qkv_kernel<<<dim3(Sc / 32, Bc * Hc), 256, 0, stream>>>(
      x, Wq, bq, Wk, bk, Wv, bv, qw, kw, vtw);
  msa_flash_kernel<<<dim3(Sc / 128, Bc * Hc), 256, 0, stream>>>(
      qw, kw, vtw, out);
}
